// GNNModel_85718957294325
// MI455X (gfx1250) — compile-verified, matrix operand order checked by reading, not run on hardware
//
#include <hip/hip_runtime.h>
#include <hip/hip_bf16.h>
#include <math.h>

#define NN       50000
#define NE       800000
#define ETOT     (NE + NN)
#define NG       64
#define NEG_SLOPE 0.2f

typedef __attribute__((ext_vector_type(16))) __bf16 v16bf;
typedef __attribute__((ext_vector_type(8)))  __bf16 v8bf;
typedef __attribute__((ext_vector_type(8)))  float  v8f;

__device__ __forceinline__ unsigned short f2bf(float f) {
    unsigned u = __float_as_uint(f);
    return (unsigned short)((u + 0x7FFFu + ((u >> 16) & 1u)) >> 16);
}

__device__ __forceinline__ void atomicMaxF(float* addr, float val) {
    if (val >= 0.0f) atomicMax((int*)addr, __float_as_int(val));
    else             atomicMin((unsigned int*)addr, __float_as_uint(val));
}

// ---------------- utility ----------------
__global__ void fill_f32(float* p, float v, int n) {
    int i = blockIdx.x * blockDim.x + threadIdx.x;
    if (i < n) p[i] = v;
}

__global__ void f32_to_bf16(const float* __restrict__ s, unsigned short* __restrict__ d, int n) {
    int i = blockIdx.x * blockDim.x + threadIdx.x;
    if (i < n) d[i] = f2bf(s[i]);
}

// Pack weight W[K,N] (f32 row-major) into WMMA B-fragment layout:
// Bp[((kt*ntiles + nt)*32 + lane)*16 + i], lane 0-15: N=lane, K-half 0; lane 16-31: K-half 1.
__global__ void pack_w(const float* __restrict__ W, unsigned short* __restrict__ Bp, int K, int N) {
    int idx = blockIdx.x * blockDim.x + threadIdx.x;
    int ntiles = N >> 4;
    int total = K * N;
    if (idx >= total) return;
    int i    = idx & 15;
    int lane = (idx >> 4) & 31;
    int t    = idx >> 9;
    int nt   = t % ntiles;
    int kt   = t / ntiles;
    int col  = lane & 15;
    int kh   = lane >> 4;
    int kk   = (i < 8) ? (kh * 8 + i) : (16 + kh * 8 + (i - 8));
    Bp[idx] = f2bf(W[(size_t)(kt * 32 + kk) * N + nt * 16 + col]);
}

// ---------------- conv1 projections (K=4, scalar) ----------------
__global__ void xlr4(const float* __restrict__ x,
                     const float* __restrict__ Wl, const float* __restrict__ bl,
                     const float* __restrict__ Wr, const float* __restrict__ br,
                     float* __restrict__ xl, float* __restrict__ xr, int Cout) {
    int idx = blockIdx.x * blockDim.x + threadIdx.x;
    if (idx >= NN * Cout) return;
    int n = idx / Cout, c = idx - n * Cout;
    const float4 xv = *reinterpret_cast<const float4*>(x + n * 4);
    float sl = bl[c], sr = br[c];
    sl += xv.x * Wl[0 * Cout + c] + xv.y * Wl[1 * Cout + c]
        + xv.z * Wl[2 * Cout + c] + xv.w * Wl[3 * Cout + c];
    sr += xv.x * Wr[0 * Cout + c] + xv.y * Wr[1 * Cout + c]
        + xv.z * Wr[2 * Cout + c] + xv.w * Wr[3 * Cout + c];
    xl[idx] = sl; xr[idx] = sr;
}

// ---------------- WMMA GEMM: C[M,N] = A[M,K](bf16) @ Bp + bias ----------------
__global__ __launch_bounds__(256)
void gemm_bf16_wmma(const unsigned short* __restrict__ A,
                    const unsigned short* __restrict__ Bp,
                    const float* __restrict__ bias,
                    float* __restrict__ C,
                    int M, int N, int K) {
    const int lane   = threadIdx.x & 31;
    const int wave   = blockIdx.x * (blockDim.x >> 5) + (threadIdx.x >> 5);
    const int ntiles = N >> 4;
    const int tiles  = (M >> 4) * ntiles;
    if (wave >= tiles) return;
    const int mt = wave / ntiles, nt = wave - mt * ntiles;
    const int row = lane & 15, kh = lane >> 4;
    const int ktiles = K >> 5;
    const unsigned short* ap = A + (size_t)(mt * 16 + row) * K + kh * 8;
    const unsigned short* bp = Bp + ((size_t)nt * 32 + lane) * 16;
    const size_t bstep = (size_t)ntiles * 32 * 16;
    v8f acc = {};
    for (int kt = 0; kt < ktiles; ++kt) {
        // WGP-scope speculative prefetch (locality 3): pull the A-tile consumed
        // two K-steps from now into near caches, not just GL2.
        __builtin_prefetch(ap + 64, 0, 3);
        v8bf a0 = *reinterpret_cast<const v8bf*>(ap);
        v8bf a1 = *reinterpret_cast<const v8bf*>(ap + 16);
        v8bf b0 = *reinterpret_cast<const v8bf*>(bp);
        v8bf b1 = *reinterpret_cast<const v8bf*>(bp + 8);
        v16bf a = __builtin_shufflevector(a0, a1, 0,1,2,3,4,5,6,7,8,9,10,11,12,13,14,15);
        v16bf b = __builtin_shufflevector(b0, b1, 0,1,2,3,4,5,6,7,8,9,10,11,12,13,14,15);
        acc = __builtin_amdgcn_wmma_f32_16x16x32_bf16(false, a, false, b,
                                                      (short)0, acc, false, false);
        ap += 32;
        bp += bstep;
    }
    const int n = nt * 16 + row;
    const float bv = bias[n];
    const int mbase = mt * 16 + kh * 8;
#pragma unroll
    for (int r = 0; r < 8; ++r)
        C[(size_t)(mbase + r) * N + n] = acc[r] + bv;
}

// ---------------- edge phase ----------------
__global__ void edge_logits(const int* __restrict__ ei,
                            const float* __restrict__ xl, const float* __restrict__ xr,
                            const float* __restrict__ att,
                            float* __restrict__ logits, float* __restrict__ mx,
                            int H, int C) {
    int idx = blockIdx.x * blockDim.x + threadIdx.x;
    if (idx >= ETOT * H) return;
    int e = idx / H, h = idx - e * H;
    int s, d;
    if (e < NE) { s = ei[e]; d = ei[NE + e]; } else { s = d = e - NE; }
    const float4* pl = reinterpret_cast<const float4*>(xl + (size_t)s * H * C + h * C);
    const float4* pr = reinterpret_cast<const float4*>(xr + (size_t)d * H * C + h * C);
    const float4* pa = reinterpret_cast<const float4*>(att + h * C);
    float acc = 0.f;
    const int C4 = C >> 2;
    for (int c = 0; c < C4; ++c) {
        float4 l = pl[c], r = pr[c], a = pa[c];
        float v0 = l.x + r.x; v0 = v0 > 0.f ? v0 : NEG_SLOPE * v0;
        float v1 = l.y + r.y; v1 = v1 > 0.f ? v1 : NEG_SLOPE * v1;
        float v2 = l.z + r.z; v2 = v2 > 0.f ? v2 : NEG_SLOPE * v2;
        float v3 = l.w + r.w; v3 = v3 > 0.f ? v3 : NEG_SLOPE * v3;
        acc += a.x * v0 + a.y * v1 + a.z * v2 + a.w * v3;
    }
    logits[idx] = acc;
    atomicMaxF(&mx[d * H + h], acc);
}

__global__ void edge_exp(const int* __restrict__ ei, float* __restrict__ logits,
                         const float* __restrict__ mx, float* __restrict__ denom, int H) {
    int idx = blockIdx.x * blockDim.x + threadIdx.x;
    if (idx >= ETOT * H) return;
    int e = idx / H, h = idx - e * H;
    int d = (e < NE) ? ei[NE + e] : (e - NE);
    float ex = __expf(logits[idx] - mx[d * H + h]);
    logits[idx] = ex;
    atomicAdd(&denom[d * H + h], ex);
}

__global__ void edge_scatter(const int* __restrict__ ei, const float* __restrict__ xl,
                             const float* __restrict__ ex, const float* __restrict__ denom,
                             float* __restrict__ out, int H, int C) {
    int idx = blockIdx.x * blockDim.x + threadIdx.x;
    int HC = H * C;
    if (idx >= ETOT * HC) return;
    int e = idx / HC, r = idx - e * HC, h = r / C;
    int s, d;
    if (e < NE) { s = ei[e]; d = ei[NE + e]; } else { s = d = e - NE; }
    float alpha = ex[e * H + h] / (denom[d * H + h] + 1e-16f);
    atomicAdd(&out[(size_t)d * HC + r], xl[(size_t)s * HC + r] * alpha);
}

__global__ void bias_act(float* __restrict__ out, const float* __restrict__ b,
                         int C, int relu) {
    int idx = blockIdx.x * blockDim.x + threadIdx.x;
    if (idx >= NN * C) return;
    int c = idx % C;
    float v = out[idx] + b[c];
    out[idx] = relu ? fmaxf(v, 0.f) : v;
}

// ---------------- pooling + MLP + LayerNorm ----------------
__global__ void pool_sum(const float* __restrict__ h, const int* __restrict__ batch,
                         float* __restrict__ pooled, float* __restrict__ cnt) {
    int idx = blockIdx.x * blockDim.x + threadIdx.x;
    if (idx >= NN * 256) return;
    int n = idx >> 8, c = idx & 255;
    int g = batch[n];
    atomicAdd(&pooled[g * 256 + c], h[idx]);
    if (c == 0) atomicAdd(&cnt[g], 1.0f);
}

__global__ void pool_div(float* __restrict__ pooled, const float* __restrict__ cnt) {
    int idx = blockIdx.x * blockDim.x + threadIdx.x;
    if (idx >= NG * 256) return;
    pooled[idx] /= fmaxf(cnt[idx >> 8], 1.0f);
}

__global__ void mlp1(const float* __restrict__ pooled, const float* __restrict__ W,
                     const float* __restrict__ b, float* __restrict__ z1) {
    int idx = blockIdx.x * blockDim.x + threadIdx.x;
    if (idx >= NG * 512) return;
    int g = idx / 512, j = idx - g * 512;
    const float* p = pooled + g * 256;
    float acc = b[j];
    for (int k = 0; k < 256; ++k) acc += p[k] * W[k * 512 + j];
    z1[idx] = fmaxf(acc, 0.f);
}

__global__ void mlp2_ln(const float* __restrict__ z1, const float* __restrict__ W,
                        const float* __restrict__ b, const float* __restrict__ lng,
                        const float* __restrict__ lnb, float* __restrict__ out) {
    __shared__ float red[256];
    int g = blockIdx.x, c = threadIdx.x;
    const float* zp = z1 + g * 512;
    float v = b[c];
    for (int k = 0; k < 512; ++k) v += zp[k] * W[k * 256 + c];
    red[c] = v; __syncthreads();
    for (int st = 128; st > 0; st >>= 1) { if (c < st) red[c] += red[c + st]; __syncthreads(); }
    float mu = red[0] / 256.f; __syncthreads();
    float dv = v - mu;
    red[c] = dv * dv; __syncthreads();
    for (int st = 128; st > 0; st >>= 1) { if (c < st) red[c] += red[c + st]; __syncthreads(); }
    float var = red[0] / 256.f;
    out[g * 256 + c] = dv * rsqrtf(var + 1e-5f) * lng[c] + lnb[c];
}

// ---------------- host orchestration ----------------
static inline int nblk(long long n) { return (int)((n + 255) / 256); }

extern "C" void kernel_launch(void* const* d_in, const int* in_sizes, int n_in,
                              void* d_out, int out_size, void* d_ws, size_t ws_size,
                              hipStream_t stream) {
    (void)in_sizes; (void)n_in; (void)out_size; (void)ws_size;
    const float* x     = (const float*)d_in[0];
    const int*   ei    = (const int*)d_in[1];
    const int*   batch = (const int*)d_in[2];
    const float *W1l=(const float*)d_in[3],  *b1l=(const float*)d_in[4];
    const float *W1r=(const float*)d_in[5],  *b1r=(const float*)d_in[6];
    const float *a1 =(const float*)d_in[7],  *c1b=(const float*)d_in[8];
    const float *W2l=(const float*)d_in[9],  *b2l=(const float*)d_in[10];
    const float *W2r=(const float*)d_in[11], *b2r=(const float*)d_in[12];
    const float *a2 =(const float*)d_in[13], *c2b=(const float*)d_in[14];
    const float *W3l=(const float*)d_in[15], *b3l=(const float*)d_in[16];
    const float *W3r=(const float*)d_in[17], *b3r=(const float*)d_in[18];
    const float *a3 =(const float*)d_in[19], *c3b=(const float*)d_in[20];
    const float *We1=(const float*)d_in[21], *be1=(const float*)d_in[22];
    const float *We2=(const float*)d_in[23], *be2=(const float*)d_in[24];
    const float *lng=(const float*)d_in[25], *lnb=(const float*)d_in[26];

    char* base = (char*)d_ws;
    size_t off = 0;
    auto alloc = [&](size_t bytes) -> void* {
        void* p = base + off;
        off += (bytes + 255) & ~(size_t)255;
        return p;
    };
    float* XL   = (float*)alloc((size_t)NN * 256 * 4);
    float* XR   = (float*)alloc((size_t)NN * 256 * 4);
    float* OUTA = (float*)alloc((size_t)NN * 256 * 4);
    float* OUTB = (float*)alloc((size_t)NN * 128 * 4);
    unsigned short* HB  = (unsigned short*)alloc((size_t)NN * 256 * 2);
    unsigned short* BPL = (unsigned short*)alloc((size_t)256 * 256 * 2);
    unsigned short* BPR = (unsigned short*)alloc((size_t)256 * 256 * 2);
    float* LOG  = (float*)alloc((size_t)ETOT * 4 * 4);
    float* MX   = (float*)alloc((size_t)NN * 4 * 4);
    float* DEN  = (float*)alloc((size_t)NN * 4 * 4);
    float* POOL = (float*)alloc((size_t)NG * 256 * 4);
    float* CNT  = (float*)alloc((size_t)NG * 4);
    float* Z1   = (float*)alloc((size_t)NG * 512 * 4);

    const int TB = 256;
    auto run_edges = [&](const float* xl, const float* xr, const float* att,
                         float* out, const float* cb, int H, int C, int relu) {
        int HC = H * C;
        fill_f32<<<nblk((long long)NN * HC), TB, 0, stream>>>(out, 0.f, NN * HC);
        fill_f32<<<nblk((long long)NN * H),  TB, 0, stream>>>(MX, -INFINITY, NN * H);
        fill_f32<<<nblk((long long)NN * H),  TB, 0, stream>>>(DEN, 0.f, NN * H);
        edge_logits<<<nblk((long long)ETOT * H), TB, 0, stream>>>(ei, xl, xr, att, LOG, MX, H, C);
        edge_exp<<<nblk((long long)ETOT * H), TB, 0, stream>>>(ei, LOG, MX, DEN, H);
        edge_scatter<<<nblk((long long)ETOT * HC), TB, 0, stream>>>(ei, xl, LOG, DEN, out, H, C);
        bias_act<<<nblk((long long)NN * HC), TB, 0, stream>>>(out, cb, HC, relu);
    };

    // conv1: 4 -> 256 (H=4,C=64), K=4 scalar projections
    xlr4<<<nblk((long long)NN * 256), TB, 0, stream>>>(x, W1l, b1l, W1r, b1r, XL, XR, 256);
    run_edges(XL, XR, a1, OUTA, c1b, 4, 64, 1);

    // conv2: 256 -> 128 (H=2,C=64), WMMA GEMMs
    f32_to_bf16<<<nblk((long long)NN * 256), TB, 0, stream>>>(OUTA, HB, NN * 256);
    pack_w<<<nblk(256LL * 128), TB, 0, stream>>>(W2l, BPL, 256, 128);
    pack_w<<<nblk(256LL * 128), TB, 0, stream>>>(W2r, BPR, 256, 128);
    {
        int tiles = (NN >> 4) * (128 >> 4);            // 25000 waves
        int blocks = (tiles + 7) / 8;
        gemm_bf16_wmma<<<blocks, 256, 0, stream>>>(HB, BPL, b2l, XL, NN, 128, 256);
        gemm_bf16_wmma<<<blocks, 256, 0, stream>>>(HB, BPR, b2r, XR, NN, 128, 256);
    }
    run_edges(XL, XR, a2, OUTB, c2b, 2, 64, 1);

    // conv3: 128 -> 256 (H=1,C=256), WMMA GEMMs
    f32_to_bf16<<<nblk((long long)NN * 128), TB, 0, stream>>>(OUTB, HB, NN * 128);
    pack_w<<<nblk(128LL * 256), TB, 0, stream>>>(W3l, BPL, 128, 256);
    pack_w<<<nblk(128LL * 256), TB, 0, stream>>>(W3r, BPR, 128, 256);
    {
        int tiles = (NN >> 4) * (256 >> 4);            // 50000 waves
        int blocks = (tiles + 7) / 8;
        gemm_bf16_wmma<<<blocks, 256, 0, stream>>>(HB, BPL, b3l, XL, NN, 256, 128);
        gemm_bf16_wmma<<<blocks, 256, 0, stream>>>(HB, BPR, b3r, XR, NN, 256, 128);
    }
    run_edges(XL, XR, a3, OUTA, c3b, 1, 256, 0);

    // global mean pool + enhancer MLP + LayerNorm
    fill_f32<<<nblk((long long)NG * 256), TB, 0, stream>>>(POOL, 0.f, NG * 256);
    fill_f32<<<1, TB, 0, stream>>>(CNT, 0.f, NG);
    pool_sum<<<nblk((long long)NN * 256), TB, 0, stream>>>(OUTA, batch, POOL, CNT);
    pool_div<<<nblk((long long)NG * 256), TB, 0, stream>>>(POOL, CNT);
    mlp1<<<nblk((long long)NG * 512), TB, 0, stream>>>(POOL, We1, be1, Z1);
    mlp2_ln<<<NG, 256, 0, stream>>>(Z1, We2, be2, lng, lnb, (float*)d_out);
}
